// S2Activation_42838003810846
// MI455X (gfx1250) — matrix operand order repaired and checked by reading, not compiled, not run on hardware
//
#include <hip/hip_runtime.h>

#define LMAX       4
#define RES_BETA   10
#define RES_ALPHA  9
#define NGRID      90          // RES_BETA * RES_ALPHA
#define NCOEF      25          // (LMAX+1)^2
#define LN_EPS     1e-6f

#define BLOCK_THREADS   256
#define WAVES_PER_BLOCK 8
#define ROWS_PER_WAVE   16
#define ROWS_PER_BLOCK  128
#define CHUNK_ELEMS     (ROWS_PER_BLOCK * NCOEF)   // 3200
#define NGRID_PAD       96     // per-wave F row stride in LDS

typedef __bf16 v16bf __attribute__((ext_vector_type(16)));
typedef float  v8f   __attribute__((ext_vector_type(8)));
typedef float  v4f   __attribute__((ext_vector_type(4)));

// ---------------------------------------------------------------------------
// Init: rebuild SH basis Y[90][25] and quadrature-weighted W[90][25] = qw*Y.
// Mirrors the reference recurrences in fp64. Runs with 90 active threads.
// ---------------------------------------------------------------------------
__global__ void s2_init_tables(float* __restrict__ Yt, float* __restrict__ Wt) {
    const int j = threadIdx.x;
    if (j >= NGRID) return;
    const int b = j / RES_ALPHA;
    const int a = j % RES_ALPHA;

    const double gl_x[RES_BETA] = {
        -0.9739065285171717, -0.8650633666889845, -0.6794095682990244,
        -0.4333953941292472, -0.1488743389816312,  0.1488743389816312,
         0.4333953941292472,  0.6794095682990244,  0.8650633666889845,
         0.9739065285171717 };
    const double gl_w[RES_BETA] = {
        0.0666713443086881, 0.1494513491505806, 0.2190863625159820,
        0.2692667193099963, 0.2955242247147529, 0.2955242247147529,
        0.2692667193099963, 0.2190863625159820, 0.1494513491505806,
        0.0666713443086881 };
    const double PI = 3.14159265358979323846;

    const double y = gl_x[b];
    const double s = sqrt(1.0 - y * y);

    double P[LMAX + 1][LMAX + 1];
    for (int l = 0; l <= LMAX; ++l)
        for (int m = 0; m <= LMAX; ++m) P[l][m] = 0.0;
    P[0][0] = 1.0;
    for (int m = 1; m <= LMAX; ++m) P[m][m] = -(2.0 * m - 1.0) * s * P[m - 1][m - 1];
    for (int m = 0; m < LMAX; ++m)  P[m + 1][m] = y * (2.0 * m + 1.0) * P[m][m];
    for (int m = 0; m <= LMAX; ++m)
        for (int l = m + 2; l <= LMAX; ++l)
            P[l][m] = ((2.0 * l - 1.0) * y * P[l - 1][m] - (l + m - 1.0) * P[l - 2][m])
                      / (double)(l - m);

    const double fact[2 * LMAX + 1] = {1, 1, 2, 6, 24, 120, 720, 5040, 40320};
    const double alpha = 2.0 * PI * (double)a / (double)RES_ALPHA;
    const double qwb   = gl_w[b] * (2.0 * PI / (double)RES_ALPHA);

    int k = 0;
    for (int l = 0; l <= LMAX; ++l) {
        for (int m = -l; m <= l; ++m, ++k) {
            const int ma = (m < 0) ? -m : m;
            const double norm = sqrt((2.0 * l + 1.0) / (4.0 * PI) *
                                     fact[l - ma] / fact[l + ma]);
            const double leg = norm * P[l][ma];
            double az;
            if (m == 0)      az = 1.0;
            else if (m > 0)  az = sqrt(2.0) * cos((double)m * alpha);
            else             az = sqrt(2.0) * sin((double)ma * alpha);
            const double v = leg * az;
            Yt[j * NCOEF + k] = (float)v;
            Wt[j * NCOEF + k] = (float)(qwb * v);
        }
    }
}

// ---------------------------------------------------------------------------
// Main fused kernel: GEMM1 (WMMA bf16) -> ReLU+LayerNorm (fp32) -> GEMM2 (WMMA)
// ---------------------------------------------------------------------------
__global__ __launch_bounds__(BLOCK_THREADS)
void s2_activation_kernel(const float* __restrict__ X,
                          const float* __restrict__ ln_scale,
                          const float* __restrict__ ln_bias,
                          const float* __restrict__ Yt,   // [90][25]
                          const float* __restrict__ Wt,   // [90][25] (qw folded)
                          float* __restrict__ out,
                          int n_rows) {
    __shared__ __align__(16) float lds_x[CHUNK_ELEMS];                      // 12.8 KB (in & out staging)
    __shared__ float lds_f[WAVES_PER_BLOCK][ROWS_PER_WAVE][NGRID_PAD];      // 48 KB

    const int tid   = threadIdx.x;
    const int wave  = tid >> 5;
    const int lane  = tid & 31;
    const int half  = lane >> 4;     // c: 0 = lanes 0-15, 1 = lanes 16-31
    const int l16   = lane & 15;

    // LayerNorm params (9 each), kept in registers
    float wsc[RES_ALPHA], wbi[RES_ALPHA];
#pragma unroll
    for (int a = 0; a < RES_ALPHA; ++a) { wsc[a] = ln_scale[a]; wbi[a] = ln_bias[a]; }

    // ---- B1 fragments: B[k][j] = Y[j][k], 6 tiles of 32x16 (K pad 25->32, j pad 90->96)
    v16bf b1[6];
#pragma unroll
    for (int n = 0; n < 6; ++n) {
        const int j = n * 16 + l16;
#pragma unroll
        for (int i = 0; i < 16; ++i) {
            const int kk = i + 16 * half;
            const float v = (kk < NCOEF && j < NGRID) ? Yt[j * NCOEF + kk] : 0.0f;
            b1[n][i] = (__bf16)v;
        }
    }
    // ---- B2 fragments: W[90][25], 3 K-tiles x 2 N-tiles of 32x16
    v16bf b2[3][2];
#pragma unroll
    for (int kt = 0; kt < 3; ++kt)
#pragma unroll
        for (int nt = 0; nt < 2; ++nt) {
            const int col = nt * 16 + l16;
#pragma unroll
            for (int i = 0; i < 16; ++i) {
                const int K = kt * 32 + i + 16 * half;
                const float v = (K < NGRID && col < NCOEF) ? Wt[K * NCOEF + col] : 0.0f;
                b2[kt][nt][i] = (__bf16)v;
            }
        }

    const int n_chunks = (n_rows + ROWS_PER_BLOCK - 1) / ROWS_PER_BLOCK;
    for (int chunk = blockIdx.x; chunk < n_chunks; chunk += gridDim.x) {
        const long R = (long)chunk * ROWS_PER_BLOCK;
        long remain = ((long)n_rows - R) * (long)NCOEF;
        if (remain > CHUNK_ELEMS) remain = CHUNK_ELEMS;
        const int n4 = (int)(remain >> 2);

        // ---- cooperative coalesced staging of X chunk into LDS (b128 loads)
        {
            const float* src  = X + R * NCOEF;
            const v4f*   src4 = (const v4f*)src;
            v4f*         dst4 = (v4f*)lds_x;
            const bool pf = (chunk + (int)gridDim.x) < n_chunks;
            const float* nsrc = src + (long)gridDim.x * (long)CHUNK_ELEMS;
            for (int q = tid; q < n4; q += BLOCK_THREADS) {
                dst4[q] = src4[q];
                if (pf) __builtin_prefetch(nsrc + 4 * q, 0, 1);   // global_prefetch_b8
            }
            for (int q = 4 * n4 + tid; q < CHUNK_ELEMS; q += BLOCK_THREADS)
                lds_x[q] = (q < remain) ? src[q] : 0.0f;
        }
        __syncthreads();

        // ---- A1 fragment (16 rows x 25 coeffs, K padded to 32), contiguous k = i + 16*half (same as B1)
        v16bf a1;
        {
            const int row = wave * ROWS_PER_WAVE + l16;
#pragma unroll
            for (int i = 0; i < 16; ++i) {
                const int kk = i + 16 * half;
                const float v = (kk < NCOEF) ? lds_x[row * NCOEF + kk] : 0.0f;
                a1[i] = (__bf16)v;
            }
        }
        __syncthreads();   // staging buffer free for reuse (output staging)

        // ---- GEMM1: F = X @ Y^T  (one K-step, 6 N-tiles)
        v8f acc1[6];
#pragma unroll
        for (int n = 0; n < 6; ++n) {
            v8f z = {0.f, 0.f, 0.f, 0.f, 0.f, 0.f, 0.f, 0.f};
            acc1[n] = __builtin_amdgcn_wmma_f32_16x16x32_bf16(
                false, a1, false, b1[n], (short)0, z, false, false);
        }

        // ---- spill F to private per-wave LDS slab (D layout: M = v+8c, N = l16)
#pragma unroll
        for (int n = 0; n < 6; ++n) {
            const int j = n * 16 + l16;
            if (j < NGRID) {
#pragma unroll
                for (int v = 0; v < 8; ++v)
                    lds_f[wave][v + 8 * half][j] = acc1[n][v];
            }
        }

        // ---- ReLU + LayerNorm over alpha (9), 160 (row,beta) tasks / wave
#pragma unroll
        for (int rep = 0; rep < (ROWS_PER_WAVE * RES_BETA) / 32; ++rep) {
            const int p    = lane + rep * 32;
            const int row  = p / RES_BETA;
            const int beta = p % RES_BETA;
            float v[RES_ALPHA];
            float mean = 0.f, m2 = 0.f;
#pragma unroll
            for (int a = 0; a < RES_ALPHA; ++a) {
                float x = lds_f[wave][row][beta * RES_ALPHA + a];
                x = fmaxf(x, 0.0f);
                v[a] = x;
                mean += x;
                m2   += x * x;
            }
            mean *= (1.0f / RES_ALPHA);
            const float var = m2 * (1.0f / RES_ALPHA) - mean * mean;
            const float inv = rsqrtf(var + LN_EPS);
#pragma unroll
            for (int a = 0; a < RES_ALPHA; ++a)
                lds_f[wave][row][beta * RES_ALPHA + a] = (v[a] - mean) * inv * wsc[a] + wbi[a];
        }

        // ---- A2 fragments: F' 16x90 (K padded to 96), 3 K-tiles
        v16bf a2[3];
#pragma unroll
        for (int kt = 0; kt < 3; ++kt)
#pragma unroll
            for (int i = 0; i < 16; ++i) {
                const int K = kt * 32 + i + 16 * half;
                const float v = (K < NGRID) ? lds_f[wave][l16][K] : 0.0f;
                a2[kt][i] = (__bf16)v;
            }

        // ---- GEMM2: out = F' @ W  (3 K-steps, 2 N-tiles, fp32 accumulate)
        v8f acc2[2];
#pragma unroll
        for (int nt = 0; nt < 2; ++nt) {
            v8f c = {0.f, 0.f, 0.f, 0.f, 0.f, 0.f, 0.f, 0.f};
#pragma unroll
            for (int kt = 0; kt < 3; ++kt)
                c = __builtin_amdgcn_wmma_f32_16x16x32_bf16(
                    false, a2[kt], false, b2[kt][nt], (short)0, c, false, false);
            acc2[nt] = c;
        }

        // ---- stage out tile into lds_x (reuse), then coalesced b128 store
#pragma unroll
        for (int nt = 0; nt < 2; ++nt) {
            const int col = nt * 16 + l16;
            if (col < NCOEF) {
#pragma unroll
                for (int v = 0; v < 8; ++v)
                    lds_x[(wave * ROWS_PER_WAVE + v + 8 * half) * NCOEF + col] = acc2[nt][v];
            }
        }
        __syncthreads();
        {
            float*     dst  = out + R * NCOEF;
            v4f*       dst4 = (v4f*)dst;
            const v4f* s4   = (const v4f*)lds_x;
            for (int q = tid; q < n4; q += BLOCK_THREADS)
                dst4[q] = s4[q];
            for (int q = 4 * n4 + tid; q < remain; q += BLOCK_THREADS)
                dst[q] = lds_x[q];
        }
        __syncthreads();   // lds_x will be overwritten by next chunk's staging
    }
}

// ---------------------------------------------------------------------------
extern "C" void kernel_launch(void* const* d_in, const int* in_sizes, int n_in,
                              void* d_out, int out_size, void* d_ws, size_t ws_size,
                              hipStream_t stream) {
    const float* X        = (const float*)d_in[0];
    const float* ln_scale = (const float*)d_in[1];
    const float* ln_bias  = (const float*)d_in[2];
    float*       out      = (float*)d_out;

    float* Yt = (float*)d_ws;                 // 90*25 fp32
    float* Wt = Yt + NGRID * NCOEF;           // 90*25 fp32

    const int n_rows = in_sizes[0] / NCOEF;

    hipLaunchKernelGGL(s2_init_tables, dim3(1), dim3(128), 0, stream, Yt, Wt);

    int n_chunks = (n_rows + ROWS_PER_BLOCK - 1) / ROWS_PER_BLOCK;
    int blocks = 2048;
    if (blocks > n_chunks) blocks = n_chunks;
    hipLaunchKernelGGL(s2_activation_kernel, dim3(blocks), dim3(BLOCK_THREADS), 0, stream,
                       X, ln_scale, ln_bias, Yt, Wt, out, n_rows);
}
